// Gat_14834817040534
// MI455X (gfx1250) — compile-verified, matrix operand order checked by reading, not run on hardware
//
#include <hip/hip_runtime.h>
#include <hip/hip_bf16.h>

// ---------- types ----------
typedef __attribute__((ext_vector_type(16))) __bf16 v16bf;
typedef __attribute__((ext_vector_type(8)))  __bf16 v8bf;
typedef __attribute__((ext_vector_type(8)))  float  v8f;

union FragAB { v16bf v; v8bf h[2]; };

// ---------- elementwise convert f32 -> bf16 ----------
__global__ void cvt_bf16_kernel(const float* __restrict__ src, __bf16* __restrict__ dst, long n) {
  long i = (long)blockIdx.x * blockDim.x + threadIdx.x;
  long stride = (long)gridDim.x * blockDim.x;
  for (; i < n; i += stride) dst[i] = (__bf16)src[i];
}

// ---------- transpose-pack W[K][NC] f32 -> Wt[NC][K] bf16 ----------
__global__ void pack_wt_kernel(const float* __restrict__ W, __bf16* __restrict__ Wt, int K, int NC) {
  int i = blockIdx.x * blockDim.x + threadIdx.x;
  if (i >= K * NC) return;
  int k = i / NC, n = i % NC;
  Wt[(size_t)n * K + k] = (__bf16)W[i];
}

// ---------- bf16 WMMA GEMM: C[M, ldc] (f32) = A[M,K] (bf16) * Bt[NCtot,K]^T ----------
// block = 256 threads = 8 waves; wave -> 16 rows x 64 cols (4 accumulator tiles)
// B tile (64 rows of Bt = 64 output cols) staged in LDS with padded stride.
template<int K>
__global__ void __launch_bounds__(256) gemm_bf16_wmma_kernel(
    const __bf16* __restrict__ A, const __bf16* __restrict__ Bt,
    float* __restrict__ C, int M, int ldc) {
  constexpr int LDSK = K + 8;                 // +16B pad per row: avoids 64-bank conflicts
  __shared__ __bf16 ldsB[64 * LDSK];
  const int tid = threadIdx.x;
  const int colBase = blockIdx.y * 64;

  // cooperative stage: 64 contiguous rows of Bt
  for (int i = tid; i < 64 * (K / 8); i += 256) {
    int r = i / (K / 8), ck = i % (K / 8);
    *(v8bf*)&ldsB[r * LDSK + ck * 8] = *(const v8bf*)&Bt[(size_t)(colBase + r) * K + ck * 8];
  }
  __syncthreads();

  const int lane = tid & 31, wave = tid >> 5;
  const int m0 = (blockIdx.x * 8 + wave) * 16;
  int mrow = m0 + (lane & 15);
  if (mrow > M - 1) mrow = M - 1;             // clamp: keep EXEC all-ones for WMMA
  const int kh = (lane >> 4) * 8;             // K sub-chunk select per ISA A-fragment layout
  const int nb = lane & 15;

  v8f acc0 = {}, acc1 = {}, acc2 = {}, acc3 = {};
  const __bf16* arow = A + (size_t)mrow * K;

  #pragma unroll
  for (int k0 = 0; k0 < K; k0 += 32) {
    FragAB a;
    a.h[0] = *(const v8bf*)(arow + k0 + kh);
    a.h[1] = *(const v8bf*)(arow + k0 + 16 + kh);

    FragAB b0, b1, b2, b3;
    const __bf16* bp0 = &ldsB[(nb +  0) * LDSK + k0];
    const __bf16* bp1 = &ldsB[(nb + 16) * LDSK + k0];
    const __bf16* bp2 = &ldsB[(nb + 32) * LDSK + k0];
    const __bf16* bp3 = &ldsB[(nb + 48) * LDSK + k0];
    b0.h[0] = *(const v8bf*)(bp0 + kh); b0.h[1] = *(const v8bf*)(bp0 + 16 + kh);
    b1.h[0] = *(const v8bf*)(bp1 + kh); b1.h[1] = *(const v8bf*)(bp1 + 16 + kh);
    b2.h[0] = *(const v8bf*)(bp2 + kh); b2.h[1] = *(const v8bf*)(bp2 + 16 + kh);
    b3.h[0] = *(const v8bf*)(bp3 + kh); b3.h[1] = *(const v8bf*)(bp3 + 16 + kh);

    acc0 = __builtin_amdgcn_wmma_f32_16x16x32_bf16(false, a.v, false, b0.v, (short)0, acc0, false, false);
    acc1 = __builtin_amdgcn_wmma_f32_16x16x32_bf16(false, a.v, false, b1.v, (short)0, acc1, false, false);
    acc2 = __builtin_amdgcn_wmma_f32_16x16x32_bf16(false, a.v, false, b2.v, (short)0, acc2, false, false);
    acc3 = __builtin_amdgcn_wmma_f32_16x16x32_bf16(false, a.v, false, b3.v, (short)0, acc3, false, false);
  }

  // C/D layout: VGPR v -> rows (m0+v) [lanes 0-15] and (m0+8+v) [lanes 16-31]; col = lane&15
  const int rbase = m0 + ((lane >> 4) << 3);
  #pragma unroll
  for (int v = 0; v < 8; ++v) {
    int row = rbase + v;
    if (row < M) {
      float* cp = C + (size_t)row * ldc + colBase;
      cp[nb]      = acc0[v];
      cp[16 + nb] = acc1[v];
      cp[32 + nb] = acc2[v];
      cp[48 + nb] = acc3[v];
    }
  }
}

// ---------- attention coefficients: es/ed[n,h] = sum_c h[n,h,c]*a[h,c] ----------
__global__ void attn_coeff_kernel(const float* __restrict__ hHC, int ldh,
                                  const float* __restrict__ as, const float* __restrict__ ad,
                                  float* __restrict__ es, float* __restrict__ ed, int N) {
  int t = blockIdx.x * blockDim.x + threadIdx.x;
  if (t >= N * 2) return;
  int n = t >> 1, h = t & 1;
  const float* hp = hHC + (size_t)n * ldh + h * 64;
  const float* sp = as + h * 64;
  const float* dp = ad + h * 64;
  float accs = 0.f, accd = 0.f;
  #pragma unroll
  for (int c = 0; c < 64; ++c) { float v = hp[c]; accs += v * sp[c]; accd += v * dp[c]; }
  es[t] = accs; ed[t] = accd;
}

// order-preserving float<->uint encoding for atomic segment-max
__device__ __forceinline__ unsigned fenc(float v) {
  unsigned u = __float_as_uint(v);
  return (u & 0x80000000u) ? ~u : (u | 0x80000000u);
}
__device__ __forceinline__ float fdec(unsigned u) {
  u = (u & 0x80000000u) ? (u & 0x7FFFFFFFu) : ~u;
  return __uint_as_float(u);
}

// ---------- edge pass 1: segment max of leaky_relu(e_src[s]+e_dst[d], 0.2) per dst ----------
__global__ void edge_max_kernel(const int* __restrict__ src, const int* __restrict__ dst,
                                const float* __restrict__ es, const float* __restrict__ ed,
                                unsigned* __restrict__ menc, int E) {
  int e = blockIdx.x * blockDim.x + threadIdx.x;
  if (e >= E) return;
  int s = src[e], d = dst[e];
  #pragma unroll
  for (int h = 0; h < 2; ++h) {
    float v = es[s * 2 + h] + ed[d * 2 + h];
    v = v > 0.f ? v : 0.2f * v;
    atomicMax(&menc[d * 2 + h], fenc(v));
  }
}

// ---------- edge pass 2: w = exp(e - m[dst]); denom[dst] += w ----------
__global__ void edge_exp_kernel(const int* __restrict__ src, const int* __restrict__ dst,
                                const float* __restrict__ es, const float* __restrict__ ed,
                                const unsigned* __restrict__ menc,
                                float* __restrict__ wbuf, float* __restrict__ denom, int E) {
  int e = blockIdx.x * blockDim.x + threadIdx.x;
  if (e >= E) return;
  int s = src[e], d = dst[e];
  #pragma unroll
  for (int h = 0; h < 2; ++h) {
    float v = es[s * 2 + h] + ed[d * 2 + h];
    v = v > 0.f ? v : 0.2f * v;
    float m = fdec(menc[d * 2 + h]);
    float w = __expf(v - m);
    wbuf[(size_t)e * 2 + h] = w;
    atomicAdd(&denom[d * 2 + h], w);
  }
}

// ---------- edge pass 3: agg[dst,h,c] += alpha * h[src,h,c]; 4 threads/edge, 32 ch each ----------
__global__ void edge_agg_kernel(const int* __restrict__ src, const int* __restrict__ dst,
                                const float* __restrict__ wbuf, const float* __restrict__ denom,
                                const float* __restrict__ hHC, int ldh,
                                float* __restrict__ agg, int E) {
  long t = (long)blockIdx.x * blockDim.x + threadIdx.x;
  int e = (int)(t >> 2);
  if (e >= E) return;
  int chunk = (int)(t & 3);
  int s = src[e], d = dst[e];
  int h = chunk >> 1;
  float alpha = wbuf[(size_t)e * 2 + h] / (denom[d * 2 + h] + 1e-16f);
  const float* hp = hHC + (size_t)s * ldh + chunk * 32;
  float* op = agg + (size_t)d * 128 + chunk * 32;
  #pragma unroll
  for (int i = 0; i < 32; ++i) atomicAdd(&op[i], alpha * hp[i]);
}

// ---------- finalize: y = mean_heads(agg) + bias + skip; accumulate col sums/sumsq ----------
__global__ void __launch_bounds__(256) finalize_kernel(
    const float* __restrict__ agg, const float* __restrict__ skip, int skipld, int skipcol,
    const float* __restrict__ bias, float* __restrict__ y, float* __restrict__ sums, int N) {
  __shared__ float sm[256];
  int c = threadIdx.x & 63, r = threadIdx.x >> 6;
  float s1 = 0.f, s2 = 0.f;
  for (int n = blockIdx.x * 4 + r; n < N; n += gridDim.x * 4) {
    float v = 0.5f * (agg[(size_t)n * 128 + c] + agg[(size_t)n * 128 + 64 + c])
            + bias[c] + skip[(size_t)n * skipld + skipcol + c];
    y[(size_t)n * 64 + c] = v;
    s1 += v; s2 += v * v;
  }
  sm[threadIdx.x] = s1; __syncthreads();
  if (r == 0) atomicAdd(&sums[c], sm[c] + sm[64 + c] + sm[128 + c] + sm[192 + c]);
  __syncthreads();
  sm[threadIdx.x] = s2; __syncthreads();
  if (r == 0) atomicAdd(&sums[64 + c], sm[c] + sm[64 + c] + sm[128 + c] + sm[192 + c]);
}

// ---------- BN stats: mu, rsigma from column sums ----------
__global__ void bnstats_kernel(const float* __restrict__ sums, float* __restrict__ bnp, int N) {
  int c = threadIdx.x;
  if (c >= 64) return;
  float mu  = sums[c] / (float)N;
  float var = sums[64 + c] / (float)N - mu * mu;
  bnp[c] = mu;
  bnp[64 + c] = rsqrtf(var + 1e-5f);
}

// ---------- BN apply + LeakyReLU(0.01) in place; optional bf16 copy for next GEMM ----------
__global__ void bnapply_kernel(float* __restrict__ y, const float* __restrict__ bnp,
                               const float* __restrict__ g, const float* __restrict__ be,
                               __bf16* __restrict__ yb, long total) {
  long i = (long)blockIdx.x * blockDim.x + threadIdx.x;
  long stride = (long)gridDim.x * blockDim.x;
  for (; i < total; i += stride) {
    int c = (int)(i & 63);
    float v = (y[i] - bnp[c]) * bnp[64 + c] * g[c] + be[c];
    v = v > 0.f ? v : 0.01f * v;
    y[i] = v;
    if (yb) yb[i] = (__bf16)v;
  }
}

// ---------- jumping knowledge max ----------
__global__ void jk_kernel(const float* __restrict__ big0, const float* __restrict__ projb,
                          const float* __restrict__ h1, const float* __restrict__ h2,
                          const float* __restrict__ h3, float* __restrict__ out, int N) {
  long i = (long)blockIdx.x * blockDim.x + threadIdx.x;
  if (i >= (long)N * 64) return;
  int c = (int)(i & 63);
  long n = i >> 6;
  float v = big0[n * 256 + 192 + c] + projb[c];   // jk0 = x @ projW + projb (fused GEMM cols 192..255)
  v = fmaxf(v, h1[i]);
  v = fmaxf(v, h2[i]);
  v = fmaxf(v, h3[i]);
  out[i] = v;
}

// ---------- host launcher ----------
extern "C" void kernel_launch(void* const* d_in, const int* in_sizes, int n_in,
                              void* d_out, int out_size, void* d_ws, size_t ws_size,
                              hipStream_t stream) {
  const float* x     = (const float*)d_in[0];
  const int*   ei    = (const int*)d_in[1];
  const float* W0    = (const float*)d_in[2];
  const float* as0   = (const float*)d_in[3];
  const float* ad0   = (const float*)d_in[4];
  const float* b0    = (const float*)d_in[5];
  const float* skip0 = (const float*)d_in[6];
  const float* g0    = (const float*)d_in[7];
  const float* be0   = (const float*)d_in[8];
  const float* W1    = (const float*)d_in[9];
  const float* as1   = (const float*)d_in[10];
  const float* ad1   = (const float*)d_in[11];
  const float* b1    = (const float*)d_in[12];
  const float* g1    = (const float*)d_in[13];
  const float* be1   = (const float*)d_in[14];
  const float* W2    = (const float*)d_in[15];
  const float* as2   = (const float*)d_in[16];
  const float* ad2   = (const float*)d_in[17];
  const float* b2    = (const float*)d_in[18];
  const float* g2    = (const float*)d_in[19];
  const float* be2   = (const float*)d_in[20];
  const float* projW = (const float*)d_in[21];
  const float* projb = (const float*)d_in[22];

  const int N = in_sizes[0] / 128;
  const int E = in_sizes[1] / 2;
  const int* src = ei;
  const int* dst = ei + E;

  // workspace carving
  char* p = (char*)d_ws;
  auto carve = [&](size_t bytes) -> char* {
    char* q = p;
    p += (bytes + 255) & ~(size_t)255;
    return q;
  };
  __bf16*   xb   = (__bf16*)  carve((size_t)N * 128 * 2);
  __bf16*   Wt0  = (__bf16*)  carve(256 * 128 * 2);
  __bf16*   Wt1  = (__bf16*)  carve(128 * 64 * 2);
  __bf16*   Wt2  = (__bf16*)  carve(128 * 64 * 2);
  float*    big0 = (float*)   carve((size_t)N * 256 * 4);   // [h0(128) | skip(64) | jk0raw(64)]
  float*    hHC  = (float*)   carve((size_t)N * 128 * 4);
  float*    es   = (float*)   carve((size_t)N * 2 * 4);
  float*    ed   = (float*)   carve((size_t)N * 2 * 4);
  unsigned* menc = (unsigned*)carve((size_t)N * 2 * 4);
  float*    den  = (float*)   carve((size_t)N * 2 * 4);
  float*    wbuf = (float*)   carve((size_t)E * 2 * 4);
  float*    agg  = (float*)   carve((size_t)N * 128 * 4);
  float*    h1   = (float*)   carve((size_t)N * 64 * 4);
  float*    h2   = (float*)   carve((size_t)N * 64 * 4);
  float*    h3   = (float*)   carve((size_t)N * 64 * 4);
  __bf16*   h1b  = (__bf16*)  carve((size_t)N * 64 * 2);
  __bf16*   h2b  = (__bf16*)  carve((size_t)N * 64 * 2);
  float*    sums = (float*)   carve(128 * 4);
  float*    bnp  = (float*)   carve(128 * 4);

  // x -> bf16
  cvt_bf16_kernel<<<2048, 256, 0, stream>>>(x, xb, (long)N * 128);

  // pack weights transposed to bf16 [NC][K]
  pack_wt_kernel<<<(128 * 128 + 255) / 256, 256, 0, stream>>>(W0,    Wt0,            128, 128);
  pack_wt_kernel<<<(128 * 64  + 255) / 256, 256, 0, stream>>>(skip0, Wt0 + 128 * 128, 128, 64);
  pack_wt_kernel<<<(128 * 64  + 255) / 256, 256, 0, stream>>>(projW, Wt0 + 192 * 128, 128, 64);
  pack_wt_kernel<<<(64 * 128  + 255) / 256, 256, 0, stream>>>(W1,    Wt1,             64, 128);
  pack_wt_kernel<<<(64 * 128  + 255) / 256, 256, 0, stream>>>(W2,    Wt2,             64, 128);

  // fused layer-0 GEMM: [N,256] = xb[N,128] @ Wt0^T
  {
    dim3 g((N + 127) / 128, 4);
    gemm_bf16_wmma_kernel<128><<<g, 256, 0, stream>>>(xb, Wt0, big0, N, 256);
  }

  const float* ASp[3] = {as0, as1, as2};
  const float* ADp[3] = {ad0, ad1, ad2};
  const float* Bp[3]  = {b0, b1, b2};
  const float* Gp[3]  = {g0, g1, g2};
  const float* BEp[3] = {be0, be1, be2};
  float*  HL[3]  = {h1, h2, h3};
  __bf16* HLB[3] = {h1b, h2b, nullptr};

  for (int l = 0; l < 3; ++l) {
    const float* hfeat;
    int ldh;
    if (l == 0) {
      hfeat = big0; ldh = 256;
    } else {
      dim3 g((N + 127) / 128, 2);
      gemm_bf16_wmma_kernel<64><<<g, 256, 0, stream>>>(HLB[l - 1], (l == 1 ? Wt1 : Wt2), hHC, N, 128);
      hfeat = hHC; ldh = 128;
    }

    attn_coeff_kernel<<<(N * 2 + 255) / 256, 256, 0, stream>>>(hfeat, ldh, ASp[l], ADp[l], es, ed, N);

    hipMemsetAsync(menc, 0, (size_t)N * 2 * 4, stream);
    hipMemsetAsync(den,  0, (size_t)N * 2 * 4, stream);
    hipMemsetAsync(agg,  0, (size_t)N * 128 * 4, stream);

    edge_max_kernel<<<(E + 255) / 256, 256, 0, stream>>>(src, dst, es, ed, menc, E);
    edge_exp_kernel<<<(E + 255) / 256, 256, 0, stream>>>(src, dst, es, ed, menc, wbuf, den, E);
    edge_agg_kernel<<<(int)(((long)E * 4 + 255) / 256), 256, 0, stream>>>(src, dst, wbuf, den, hfeat, ldh, agg, E);

    hipMemsetAsync(sums, 0, 128 * 4, stream);
    const float* skipsrc;
    int skipld, skipcol;
    if (l == 0) { skipsrc = big0;      skipld = 256; skipcol = 128; }
    else        { skipsrc = HL[l - 1]; skipld = 64;  skipcol = 0;   }
    finalize_kernel<<<1024, 256, 0, stream>>>(agg, skipsrc, skipld, skipcol, Bp[l], HL[l], sums, N);
    bnstats_kernel<<<1, 64, 0, stream>>>(sums, bnp, N);
    bnapply_kernel<<<2048, 256, 0, stream>>>(HL[l], bnp, Gp[l], BEp[l], HLB[l], (long)N * 64);
  }

  jk_kernel<<<(int)(((long)N * 64 + 255) / 256), 256, 0, stream>>>(big0, projb, h1, h2, h3, (float*)d_out, N);
}
